// LocalUnmergeBlock_1211180777859
// MI455X (gfx1250) — matrix operand order repaired and checked by reading, not compiled, not run on hardware
//
#include <hip/hip_runtime.h>

#define DD   768
#define HH   12
#define HDIM 64
#define WIN  128
#define BB   4
#define LL   8190
#define LP   8192
#define MROWS (BB * LP)   // 32768

typedef __bf16 bf16;
typedef __attribute__((ext_vector_type(16))) __bf16 bf16x16;
typedef __attribute__((ext_vector_type(8)))  float  f32x8;
typedef __attribute__((ext_vector_type(4)))  unsigned int u32x4;
typedef __attribute__((ext_vector_type(8)))  int i32x8;
typedef __attribute__((ext_vector_type(4)))  int i32x4;

#if __has_builtin(__builtin_amdgcn_tensor_load_to_lds)
#define HAVE_TDM 1
#endif

// fp32 -> bf16 round-to-nearest-even
__device__ __forceinline__ unsigned short f2bf(float f) {
  union { float f; unsigned int u; } c; c.f = f;
  unsigned int u = c.u;
  u += 0x7fffu + ((u >> 16) & 1u);
  return (unsigned short)(u >> 16);
}

union Frag {
  bf16x16 v;
  unsigned short u[16];
  uint4 q[2];
};

__device__ __forceinline__ f32x8 wmma_bf16(const Frag& a, const Frag& b, f32x8 c) {
  return __builtin_amdgcn_wmma_f32_16x16x32_bf16(
      false, a.v, false, b.v, (short)0, c, false, false);
}

// LDS byte offset of a __shared__ object (flat LDS addr = {SHARED_BASE, off32})
__device__ __forceinline__ unsigned int lds_off(const void* p) {
  return (unsigned int)(unsigned long long)(uintptr_t)p;
}

// CDNA5 async global->LDS copy, 16 bytes per lane (ASYNCcnt-tracked)
__device__ __forceinline__ void async_copy_b128(unsigned int ldsaddr, const void* g) {
  unsigned long long ga = (unsigned long long)(uintptr_t)g;
  asm volatile("global_load_async_to_lds_b128 %0, %1, off"
               :: "v"(ldsaddr), "v"(ga) : "memory");
}

__device__ __forceinline__ void wait_async0() {
  asm volatile("s_wait_asynccnt 0x0" ::: "memory");
}

#ifdef HAVE_TDM
// TDM 2D tile load: tile_cols x tile_rows of 2-byte elements, row stride in
// elements. Descriptor bitfields per CDNA5 ISA 8.3 (group0) / 8.4 (group1).
__device__ __forceinline__ void tdm_load_2d(unsigned int lds_addr, const void* gp,
                                            unsigned tile_cols, unsigned tile_rows,
                                            unsigned long long stride_elems,
                                            unsigned long long tensor_rows) {
  unsigned long long ga = (unsigned long long)(uintptr_t)gp;
  u32x4 g0;
  g0[0] = 1u;                                           // count=1, user mode
  g0[1] = lds_addr;                                     // lds_addr (bytes)
  g0[2] = (unsigned int)(ga & 0xffffffffu);             // global_addr[31:0]
  g0[3] = (unsigned int)((ga >> 32) & 0x01ffffffu)      // global_addr[56:32]
        | (2u << 30);                                   // type=2 ("image")
  const unsigned dim0 = (unsigned)stride_elems;         // tensor_dim0 (elems)
  i32x8 g1;
  g1[0] = (int)(1u << 16);                              // data_size=1 (2 bytes)
  g1[1] = (int)((dim0 & 0xffffu) << 16);                // tensor_dim0[15:0]
  g1[2] = (int)(((dim0 >> 16) & 0xffffu)                // tensor_dim0[31:16]
        | (((unsigned)(tensor_rows & 0xffffu)) << 16)); // tensor_dim1[15:0]
  g1[3] = (int)(((unsigned)((tensor_rows >> 16) & 0xffffu)) // tensor_dim1[31:16]
        | (tile_cols << 16));                           // tile_dim0
  g1[4] = (int)tile_rows;                               // tile_dim1 (tile_dim2=0)
  g1[5] = (int)(unsigned)(stride_elems & 0xffffffffu);  // tensor_dim0_stride lo
  g1[6] = (int)(unsigned)((stride_elems >> 32) & 0xffffu); // stride hi16
  g1[7] = 0;
  i32x4 z4 = {0, 0, 0, 0};
  i32x8 z8 = {0, 0, 0, 0, 0, 0, 0, 0};
  __builtin_amdgcn_tensor_load_to_lds(g0, g1, z4, z4, z8, 0);
}
#endif

// ---------------------------------------------------------------------------
// Kernel 1a: fp32 X (B,L,D) -> bf16 Xb (B*LP, D) with zero-padded rows
// ---------------------------------------------------------------------------
__global__ __launch_bounds__(256)
void k_convert(const float* __restrict__ x, unsigned short* __restrict__ xb) {
  long long idx = ((long long)blockIdx.x * blockDim.x + threadIdx.x) * 4;
  long long row = idx / DD;
  int col = (int)(idx % DD);
  int b = (int)(row >> 13);          // / LP
  int l = (int)(row & (LP - 1));
  float4 f = make_float4(0.f, 0.f, 0.f, 0.f);
  if (l < LL) {
    const float* src = x + ((long long)(b * (long long)LL + l) * DD + col);
    f = *(const float4*)src;
  }
  unsigned short o[4] = {f2bf(f.x), f2bf(f.y), f2bf(f.z), f2bf(f.w)};
  *(uint2*)(xb + idx) = *(const uint2*)o;
}

// ---------------------------------------------------------------------------
// Kernel 1b: convert + transpose the 4 weight matrices once:
//   WT[n][k] = bf16(W[k][n]),  so GEMM B-tiles become raw DMA copies.
// ---------------------------------------------------------------------------
__global__ __launch_bounds__(256)
void k_wconv(const float* __restrict__ W0, const float* __restrict__ W1,
             const float* __restrict__ W2, const float* __restrict__ W3,
             unsigned short* __restrict__ dst) {
  const float* W = (blockIdx.y == 0) ? W0 : (blockIdx.y == 1) ? W1
                 : (blockIdx.y == 2) ? W2 : W3;
  unsigned short* WT = dst + (size_t)blockIdx.y * DD * DD;
  int idx = (blockIdx.x * 256 + threadIdx.x) * 4;   // over 768*768
  int k = idx / DD;
  int n = idx % DD;
  float4 w4 = *(const float4*)(W + idx);
  WT[(n + 0) * DD + k] = f2bf(w4.x);
  WT[(n + 1) * DD + k] = f2bf(w4.y);
  WT[(n + 2) * DD + k] = f2bf(w4.z);
  WT[(n + 3) * DD + k] = f2bf(w4.w);
}

// ---------------------------------------------------------------------------
// GEMM main loop: block tile 128x128, 8 waves (2x4), wave 64x32, K-chunk 32.
// Double-buffered LDS. Tiles staged by the Tensor Data Mover (one descriptor
// per tile, issued by wave 0, TENSORcnt-tracked); per-lane async copies as
// fallback when the TDM builtin is unavailable. DMA of tile kt+1 overlaps
// WMMA compute of tile kt; one wait + one barrier per K-iteration.
// ---------------------------------------------------------------------------
__device__ __forceinline__ void stage_tiles(
    const unsigned short* __restrict__ Abf, const unsigned short* __restrict__ WbT,
    unsigned int a_base, unsigned int b_base, int mblk, int nblk, int k0,
    int tid, int wv) {
#ifdef HAVE_TDM
  (void)tid;
  if (wv == 0) {
    tdm_load_2d(a_base, Abf + (long long)mblk * DD + k0, 32, 128, DD, MROWS);
    tdm_load_2d(b_base, WbT + (long long)nblk * DD + k0, 32, 128, DD, DD);
  }
#else
  (void)wv;
  #pragma unroll
  for (int i = 0; i < 2; ++i) {
    int c  = tid + i * 256;     // 512 b128 chunks per tile
    int r  = c >> 2;
    int cc = (c & 3) * 8;
    async_copy_b128(a_base + (unsigned)(r * 32 + cc) * 2,
                    Abf + (long long)(mblk + r) * DD + k0 + cc);
    async_copy_b128(b_base + (unsigned)(r * 32 + cc) * 2,
                    WbT + (long long)(nblk + r) * DD + k0 + cc);
  }
#endif
}

__device__ __forceinline__ void stage_wait(int wv) {
#ifdef HAVE_TDM
  if (wv == 0) __builtin_amdgcn_s_wait_tensorcnt(0);
#else
  (void)wv;
  wait_async0();
#endif
  __syncthreads();
}

__device__ __forceinline__ void gemm_mainloop(
    const unsigned short* __restrict__ Abf, const unsigned short* __restrict__ WbT,
    unsigned short (*Asm)[128 * 32], unsigned short (*BsT)[128 * 32],
    int mblk, int nblk, int tid, int lane, int wm, int wn,
    f32x8 acc[4][2]) {
  const int wv = tid >> 5;
  const unsigned int ab[2] = { lds_off(Asm[0]), lds_off(Asm[1]) };
  const unsigned int bb[2] = { lds_off(BsT[0]), lds_off(BsT[1]) };

  stage_tiles(Abf, WbT, ab[0], bb[0], mblk, nblk, 0, tid, wv);

  for (int kt = 0; kt < DD / 32; ++kt) {
    const int cur = kt & 1;
    stage_wait(wv);
    if (kt + 1 < DD / 32)
      stage_tiles(Abf, WbT, ab[cur ^ 1], bb[cur ^ 1], mblk, nblk, (kt + 1) * 32, tid, wv);

    const unsigned short* As = Asm[cur];
    const unsigned short* Bs = BsT[cur];
    // batch all fragment loads, then all WMMAs
    Frag bfr[2], a4[4];
    #pragma unroll
    for (int ns = 0; ns < 2; ++ns) {
      int ncol = wn + ns * 16 + (lane & 15);
      int kb   = (lane >> 4) * 8;
      bfr[ns].q[0] = *(const uint4*)(Bs + ncol * 32 + kb);
      bfr[ns].q[1] = *(const uint4*)(Bs + ncol * 32 + kb + 16);
    }
    #pragma unroll
    for (int ms = 0; ms < 4; ++ms) {
      int row = wm + ms * 16 + (lane & 15);
      int kb  = (lane >> 4) * 8;
      a4[ms].q[0] = *(const uint4*)(As + row * 32 + kb);
      a4[ms].q[1] = *(const uint4*)(As + row * 32 + 16 + kb);
    }
    #pragma unroll
    for (int ms = 0; ms < 4; ++ms)
      #pragma unroll
      for (int ns = 0; ns < 2; ++ns)
        acc[ms][ns] = wmma_bf16(a4[ms], bfr[ns], acc[ms][ns]);
  }
}

// ---------------------------------------------------------------------------
// Kernel 2: C_bf16 = A_bf16 @ W + bias.  TRANSPOSED=true stores C^T
// (col-major, stride MROWS) -- used for V so attention can DMA V^T tiles.
// ---------------------------------------------------------------------------
template <bool TRANSPOSED>
__global__ __launch_bounds__(256)
void k_gemm_qkv(const unsigned short* __restrict__ Xb,
                const unsigned short* __restrict__ WbT,
                const float* __restrict__ bias,
                unsigned short* __restrict__ Cout) {
  __shared__ unsigned short Asm[2][128 * 32];
  __shared__ unsigned short BsT[2][128 * 32];

  const int tid  = threadIdx.x;
  const int wv   = tid >> 5;
  const int lane = tid & 31;
  const int mblk = blockIdx.x * 128;
  const int nblk = blockIdx.y * 128;
  const int wm = (wv >> 2) * 64;
  const int wn = (wv & 3) * 32;

  f32x8 acc[4][2];
  #pragma unroll
  for (int i = 0; i < 4; ++i)
    #pragma unroll
    for (int j = 0; j < 2; ++j)
      acc[i][j] = f32x8{0.f, 0.f, 0.f, 0.f, 0.f, 0.f, 0.f, 0.f};

  gemm_mainloop(Xb, WbT, Asm, BsT, mblk, nblk, tid, lane, wm, wn, acc);

  #pragma unroll
  for (int ms = 0; ms < 4; ++ms)
    #pragma unroll
    for (int ns = 0; ns < 2; ++ns) {
      int colbase = nblk + wn + ns * 16 + (lane & 15);
      float bi = bias[colbase];
      #pragma unroll
      for (int e = 0; e < 8; ++e) {
        int r = mblk + wm + ms * 16 + e + ((lane >> 4) << 3);
        unsigned short val = f2bf(acc[ms][ns][e] + bi);
        if (TRANSPOSED)
          Cout[(long long)colbase * MROWS + r] = val;
        else
          Cout[(long long)r * DD + colbase] = val;
      }
    }
}

// ---------------------------------------------------------------------------
// Kernel 3: per-(batch, window, head) attention. Q,K row-major; V^T col-major.
// All LDS staging via per-lane async global->LDS.
// ---------------------------------------------------------------------------
__global__ __launch_bounds__(256)
void k_attn(const unsigned short* __restrict__ Qb,
            const unsigned short* __restrict__ Kb,
            const unsigned short* __restrict__ Vt,
            unsigned short* __restrict__ Ob) {
  __shared__ unsigned short Ks[WIN * HDIM];    // [kk][d]   128x64
  __shared__ unsigned short VsT[HDIM * WIN];   // [d][kk]   64x128
  __shared__ unsigned short Ps[8][16 * WIN];   // per-wave 16x128 probs

  const int tid  = threadIdx.x;
  const int wv   = tid >> 5;
  const int lane = tid & 31;
  const int h    = blockIdx.x % HH;
  const int wnd  = (blockIdx.x / HH) & 63;
  const int b    = blockIdx.x / (HH * 64);
  const long long rb = (long long)(b * 64 + wnd) * WIN;  // global row base
  const int cb = h * HDIM;

  const unsigned int ks_base = lds_off(Ks);
  #pragma unroll
  for (int i = 0; i < 4; ++i) {
    int c  = tid + i * 256;
    int r  = c >> 3;
    int cc = (c & 7) * 8;
    async_copy_b128(ks_base + (unsigned)(r * HDIM + cc) * 2,
                    Kb + (rb + r) * DD + cb + cc);
  }
  const unsigned int vs_base = lds_off(VsT);
  #pragma unroll
  for (int i = 0; i < 4; ++i) {
    int c  = tid + i * 256;
    int d  = c >> 4;
    int kk = (c & 15) * 8;
    async_copy_b128(vs_base + (unsigned)(d * WIN + kk) * 2,
                    Vt + (long long)(cb + d) * MROWS + rb + kk);
  }
  wait_async0();
  __syncthreads();

  Frag qa[2];
  {
    const unsigned short* qp = Qb + (rb + wv * 16 + (lane & 15)) * DD + cb;
    int kb = (lane >> 4) * 8;
    #pragma unroll
    for (int c = 0; c < 2; ++c) {
      qa[c].q[0] = *(const uint4*)(qp + c * 32 + kb);
      qa[c].q[1] = *(const uint4*)(qp + c * 32 + 16 + kb);
    }
  }

  f32x8 sc[8];
  #pragma unroll
  for (int nt = 0; nt < 8; ++nt) {
    f32x8 a = f32x8{0.f, 0.f, 0.f, 0.f, 0.f, 0.f, 0.f, 0.f};
    #pragma unroll
    for (int kc = 0; kc < 2; ++kc) {
      Frag bfr;
      int kk = nt * 16 + (lane & 15);
      int db = kc * 32 + (lane >> 4) * 8;
      bfr.q[0] = *(const uint4*)(Ks + kk * HDIM + db);
      bfr.q[1] = *(const uint4*)(Ks + kk * HDIM + db + 16);
      a = wmma_bf16(qa[kc], bfr, a);
    }
    sc[nt] = a;
  }

  const float scale = 0.125f;  // 1/sqrt(64)
  float rinv[8];
  #pragma unroll
  for (int v = 0; v < 8; ++v) {
    float pm = -3.0e38f;
    #pragma unroll
    for (int nt = 0; nt < 8; ++nt) pm = fmaxf(pm, sc[nt][v]);
    pm *= scale;
    #pragma unroll
    for (int off = 8; off; off >>= 1) pm = fmaxf(pm, __shfl_xor(pm, off, 32));
    float s = 0.f;
    #pragma unroll
    for (int nt = 0; nt < 8; ++nt) {
      float p = __expf(sc[nt][v] * scale - pm);
      sc[nt][v] = p;
      s += p;
    }
    #pragma unroll
    for (int off = 8; off; off >>= 1) s += __shfl_xor(s, off, 32);
    rinv[v] = 1.0f / s;
  }

  unsigned short* Pw = Ps[wv];
  #pragma unroll
  for (int nt = 0; nt < 8; ++nt)
    #pragma unroll
    for (int v = 0; v < 8; ++v) {
      int r  = v + ((lane >> 4) << 3);
      int cc = nt * 16 + (lane & 15);
      Pw[r * WIN + cc] = f2bf(sc[nt][v] * rinv[v]);
    }
  __syncthreads();

  f32x8 oacc[4];
  #pragma unroll
  for (int nt = 0; nt < 4; ++nt) oacc[nt] = f32x8{0.f, 0.f, 0.f, 0.f, 0.f, 0.f, 0.f, 0.f};
  #pragma unroll
  for (int kc = 0; kc < 4; ++kc) {
    Frag pa;
    int row = (lane & 15);
    int kb  = kc * 32 + (lane >> 4) * 8;
    pa.q[0] = *(const uint4*)(Pw + row * WIN + kb);
    pa.q[1] = *(const uint4*)(Pw + row * WIN + kb + 16);
    #pragma unroll
    for (int nt = 0; nt < 4; ++nt) {
      Frag bfr;
      int n  = nt * 16 + (lane & 15);
      int kk = kc * 32 + (lane >> 4) * 8;
      bfr.q[0] = *(const uint4*)(VsT + n * WIN + kk);
      bfr.q[1] = *(const uint4*)(VsT + n * WIN + kk + 16);
      oacc[nt] = wmma_bf16(pa, bfr, oacc[nt]);
    }
  }

  #pragma unroll
  for (int nt = 0; nt < 4; ++nt)
    #pragma unroll
    for (int e = 0; e < 8; ++e) {
      int r  = wv * 16 + e + ((lane >> 4) << 3);
      int cc = nt * 16 + (lane & 15);
      Ob[(rb + r) * DD + cb + cc] = f2bf(oacc[nt][e]);
    }
}

// ---------------------------------------------------------------------------
// Kernel 4: Y_f32 = O_bf16 @ Wo + bo + X (residual)
// ---------------------------------------------------------------------------
__global__ __launch_bounds__(256)
void k_gemm_out(const unsigned short* __restrict__ Ob,
                const unsigned short* __restrict__ WbT,
                const float* __restrict__ bias,
                const float* __restrict__ Xin,
                float* __restrict__ Y) {
  __shared__ unsigned short Asm[2][128 * 32];
  __shared__ unsigned short BsT[2][128 * 32];

  const int tid  = threadIdx.x;
  const int wv   = tid >> 5;
  const int lane = tid & 31;
  const int mblk = blockIdx.x * 128;
  const int nblk = blockIdx.y * 128;
  const int wm = (wv >> 2) * 64;
  const int wn = (wv & 3) * 32;

  f32x8 acc[4][2];
  #pragma unroll
  for (int i = 0; i < 4; ++i)
    #pragma unroll
    for (int j = 0; j < 2; ++j)
      acc[i][j] = f32x8{0.f, 0.f, 0.f, 0.f, 0.f, 0.f, 0.f, 0.f};

  gemm_mainloop(Ob, WbT, Asm, BsT, mblk, nblk, tid, lane, wm, wn, acc);

  #pragma unroll
  for (int ms = 0; ms < 4; ++ms)
    #pragma unroll
    for (int ns = 0; ns < 2; ++ns) {
      int colbase = nblk + wn + ns * 16 + (lane & 15);
      float bi = bias[colbase];
      #pragma unroll
      for (int e = 0; e < 8; ++e) {
        int r = mblk + wm + ms * 16 + e + ((lane >> 4) << 3);
        int bb = r >> 13, l = r & (LP - 1);
        float res = 0.f;
        if (l < LL) res = Xin[((long long)bb * LL + l) * DD + colbase];
        Y[(long long)r * DD + colbase] = acc[ms][ns][e] + bi + res;
      }
    }
}

// ---------------------------------------------------------------------------
// Kernel 5: row-wise LayerNorm -> d_out (only valid rows)
// ---------------------------------------------------------------------------
__global__ __launch_bounds__(256)
void k_ln(const float* __restrict__ Y, const float* __restrict__ g,
          const float* __restrict__ be, float* __restrict__ out) {
  __shared__ float rbuf[8];
  const int row = blockIdx.x;            // 0 .. B*L-1
  const int b = row / LL, l = row % LL;
  const float* y = Y + ((long long)(b * LP + l)) * DD;
  float* o = out + (long long)row * DD;
  const int tid = threadIdx.x;

  float v3[3];
  float s = 0.f;
  #pragma unroll
  for (int i = 0; i < 3; ++i) { v3[i] = y[tid + i * 256]; s += v3[i]; }
  #pragma unroll
  for (int off = 16; off; off >>= 1) s += __shfl_xor(s, off, 32);
  if ((tid & 31) == 0) rbuf[tid >> 5] = s;
  __syncthreads();
  float mean = 0.f;
  #pragma unroll
  for (int i = 0; i < 8; ++i) mean += rbuf[i];
  mean *= (1.0f / 768.0f);
  __syncthreads();

  float q = 0.f;
  #pragma unroll
  for (int i = 0; i < 3; ++i) { float d = v3[i] - mean; q += d * d; }
  #pragma unroll
  for (int off = 16; off; off >>= 1) q += __shfl_xor(q, off, 32);
  if ((tid & 31) == 0) rbuf[tid >> 5] = q;
  __syncthreads();
  float var = 0.f;
  #pragma unroll
  for (int i = 0; i < 8; ++i) var += rbuf[i];
  var *= (1.0f / 768.0f);
  float inv = rsqrtf(var + 1e-5f);

  #pragma unroll
  for (int i = 0; i < 3; ++i) {
    int c = tid + i * 256;
    o[c] = (v3[i] - mean) * inv * g[c] + be[c];
  }
}

// ---------------------------------------------------------------------------
extern "C" void kernel_launch(void* const* d_in, const int* in_sizes, int n_in,
                              void* d_out, int out_size, void* d_ws, size_t ws_size,
                              hipStream_t stream) {
  (void)in_sizes; (void)n_in; (void)out_size; (void)ws_size;
  const float* x   = (const float*)d_in[0];
  const float* Wq  = (const float*)d_in[1];
  const float* bq  = (const float*)d_in[2];
  const float* Wk  = (const float*)d_in[3];
  const float* bk  = (const float*)d_in[4];
  const float* Wv  = (const float*)d_in[5];
  const float* bv  = (const float*)d_in[6];
  const float* Wo  = (const float*)d_in[7];
  const float* bo  = (const float*)d_in[8];
  const float* lng = (const float*)d_in[9];
  const float* lnb = (const float*)d_in[10];

  const long long MB = (long long)MROWS * DD * 2;  // 50331648 B per bf16 matrix
  const long long WB = (long long)DD * DD * 2;     // 1179648 B per bf16 weight
  char* ws = (char*)d_ws;
  unsigned short* Xb  = (unsigned short*)(ws);
  unsigned short* Qb  = (unsigned short*)(ws + 1 * MB);
  unsigned short* Kb  = (unsigned short*)(ws + 2 * MB);
  unsigned short* Vt  = (unsigned short*)(ws + 3 * MB);   // transposed: [col][row]
  unsigned short* Ob  = (unsigned short*)(ws + 4 * MB);
  unsigned short* WqT = (unsigned short*)(ws + 5 * MB);
  unsigned short* WkT = (unsigned short*)(ws + 5 * MB + 1 * WB);
  unsigned short* WvT = (unsigned short*)(ws + 5 * MB + 2 * WB);
  unsigned short* WoT = (unsigned short*)(ws + 5 * MB + 3 * WB);
  float* Y = (float*)(ws);  // aliases Xb+Qb, both dead before k_gemm_out runs

  k_convert<<<(MROWS * DD / 4) / 256, 256, 0, stream>>>(x, Xb);
  k_wconv<<<dim3(DD * DD / 1024, 4), 256, 0, stream>>>(Wq, Wk, Wv, Wo, WqT);

  dim3 g2(MROWS / 128, DD / 128);
  k_gemm_qkv<false><<<g2, 256, 0, stream>>>(Xb, WqT, bq, Qb);
  k_gemm_qkv<false><<<g2, 256, 0, stream>>>(Xb, WkT, bk, Kb);
  k_gemm_qkv<true ><<<g2, 256, 0, stream>>>(Xb, WvT, bv, Vt);

  k_attn<<<BB * 64 * HH, 256, 0, stream>>>(Qb, Kb, Vt, Ob);

  k_gemm_out<<<g2, 256, 0, stream>>>(Ob, WoT, bo, x, Y);

  k_ln<<<BB * LL, 256, 0, stream>>>(Y, lng, lnb, (float*)d_out);
}
